// SCLSTMLayer_18906446037642
// MI455X (gfx1250) — compile-verified, matrix operand order checked by reading, not run on hardware
//
#include <hip/hip_runtime.h>
#include <hip/hip_bf16.h>
#include <math.h>

// ---------------------------------------------------------------------------
// SC-LSTM on gfx1250 (MI455X). wave32, WMMA bf16 16x16x32, f32 accumulate.
// Persistent recurrent kernel: 64 WGs x 8 waves, K split across wave pairs.
// ---------------------------------------------------------------------------

typedef __attribute__((ext_vector_type(16))) __bf16 v16bf;
typedef __attribute__((ext_vector_type(8)))  __bf16 v8bf;
typedef __attribute__((ext_vector_type(8)))  float  v8f;

#define T_STEPS 256
#define BATCH   64
#define IDIM    512
#define HDIM    1024
#define H4      4096
#define DDIM    100
#define DPAD    128
#define NWG     64   // workgroups in the persistent recurrent kernel

__device__ __forceinline__ v8f vzero8() {
    v8f z;
#pragma unroll
    for (int i = 0; i < 8; ++i) z[i] = 0.0f;
    return z;
}

__device__ __forceinline__ float sigmf(float x) {
    return 1.0f / (1.0f + __expf(-x));
}

// Load a 16x32 bf16 fragment (A-side: row0 = M base; B-side: row0 = N base,
// from a pre-transposed [n][k] matrix). Per CDNA5 ISA 16-bit operand layout:
//   lanes 0-15 : rows 0-15, K = k0+0..7  and k0+16..23
//   lanes 16-31: rows 0-15, K = k0+8..15 and k0+24..31
__device__ __forceinline__ v16bf load_frag(const __bf16* base, int ld, int row0, int k0) {
    int lane = threadIdx.x & 31;
    int r    = row0 + (lane & 15);
    int kb   = k0 + ((lane >> 4) << 3);
    const __bf16* p = base + (size_t)r * (size_t)ld + kb;
    v8bf lo = *reinterpret_cast<const v8bf*>(p);
    v8bf hi = *reinterpret_cast<const v8bf*>(p + 16);
    v16bf out;
#pragma unroll
    for (int i = 0; i < 8; ++i) { out[i] = lo[i]; out[i + 8] = hi[i]; }
    return out;
}

// ---------------------------------------------------------------------------
// Prep kernels: f32->bf16 convert, transpose-convert with zero pad, state init
// ---------------------------------------------------------------------------

__global__ void cvt_bf16(const float* __restrict__ s, __bf16* __restrict__ d, int n) {
    int i = blockIdx.x * blockDim.x + threadIdx.x;
    if (i < n) d[i] = (__bf16)s[i];
}

// dst is [DR][DC] row-major; dst[c][r] = src[r][c] (zero outside RxC)
__global__ void transpose_cvt(const float* __restrict__ src, __bf16* __restrict__ dst,
                              int R, int C, int DR, int DC) {
    int idx = blockIdx.x * blockDim.x + threadIdx.x;
    if (idx >= DR * DC) return;
    int c = idx / DC;
    int r = idx - c * DC;
    float v = (c < C && r < R) ? src[(size_t)r * C + c] : 0.0f;
    dst[idx] = (__bf16)v;
}

__global__ void init_state(const float* __restrict__ h0, const float* __restrict__ d0,
                           __bf16* __restrict__ hbf, __bf16* __restrict__ dbf,
                           float* __restrict__ dfp, unsigned* __restrict__ bar) {
    int idx = blockIdx.x * blockDim.x + threadIdx.x;
    if (idx < 2) bar[idx] = 0u;
    if (idx < BATCH * HDIM) hbf[idx] = (__bf16)h0[idx];
    if (idx < BATCH * DPAD) {
        int m = idx / DPAD, c = idx - m * DPAD;
        float v = (c < DDIM) ? d0[m * DDIM + c] : 0.0f;
        dfp[idx] = v;
        dbf[idx] = (__bf16)v;
    }
}

// ---------------------------------------------------------------------------
// Parallel GEMM: C[M][N](f32) = A[M][K](bf16,row) * BT[N][K](bf16) (+ bias[n])
// block = 128 threads (4 waves). Each wave: 16 N-cols x 64 M-rows.
// ---------------------------------------------------------------------------

__global__ void __launch_bounds__(128) gemm_bf16_tn(
    const __bf16* __restrict__ A, const __bf16* __restrict__ BT,
    const float* __restrict__ bias, float* __restrict__ C,
    int M, int N, int K)
{
    int w    = threadIdx.x >> 5;
    int lane = threadIdx.x & 31;
    int n0   = blockIdx.x * 64 + w * 16;
    int m0   = blockIdx.y * 64;
    int mofs = (lane >> 4) << 3;
    int col  = lane & 15;

    v8f acc[4];
#pragma unroll
    for (int mi = 0; mi < 4; ++mi) acc[mi] = vzero8();

#pragma unroll 4
    for (int k0 = 0; k0 < K; k0 += 32) {
        v16bf bf = load_frag(BT, K, n0, k0);
#pragma unroll
        for (int mi = 0; mi < 4; ++mi) {
            v16bf af = load_frag(A, K, m0 + mi * 16, k0);
            acc[mi] = __builtin_amdgcn_wmma_f32_16x16x32_bf16(
                false, af, false, bf, (short)0, acc[mi], false, false);
        }
    }

    float bv = bias ? bias[n0 + col] : 0.0f;
#pragma unroll
    for (int mi = 0; mi < 4; ++mi) {
#pragma unroll
        for (int v = 0; v < 8; ++v) {
            int m = m0 + mi * 16 + v + mofs;
            C[(size_t)m * N + n0 + col] = acc[mi][v] + bv;
        }
    }
}

// ---------------------------------------------------------------------------
// Device-wide barrier for the persistent recurrent kernel
// ---------------------------------------------------------------------------

__device__ __forceinline__ void grid_barrier(unsigned* cnt, unsigned* gen) {
    __threadfence();
    __syncthreads();
    if (threadIdx.x == 0) {
        unsigned g = __hip_atomic_load(gen, __ATOMIC_RELAXED, __HIP_MEMORY_SCOPE_AGENT);
        unsigned a = __hip_atomic_fetch_add(cnt, 1u, __ATOMIC_ACQ_REL, __HIP_MEMORY_SCOPE_AGENT);
        if (a == (unsigned)(NWG - 1)) {
            __hip_atomic_store(cnt, 0u, __ATOMIC_RELAXED, __HIP_MEMORY_SCOPE_AGENT);
            __hip_atomic_store(gen, g + 1u, __ATOMIC_RELEASE, __HIP_MEMORY_SCOPE_AGENT);
        } else {
            while (__hip_atomic_load(gen, __ATOMIC_ACQUIRE, __HIP_MEMORY_SCOPE_AGENT) == g) {
                __builtin_amdgcn_s_sleep(2);
            }
        }
        __threadfence();
    }
    __syncthreads();
}

// ---------------------------------------------------------------------------
// Persistent recurrent kernel. 64 WGs x 256 threads (8 waves). WG j owns H
// columns [j*16, j*16+16). Wave w: gate type (w&3), K-half (w>>2) -> two
// waves per SIMD, halved sequential wmma chains, LDS partial reduction.
// WGs 0..6 also own 16 columns each of the (128-padded) dialogue-act vector.
// ---------------------------------------------------------------------------

__global__ void __launch_bounds__(256) sclstm_recur(
    const float*  __restrict__ xwx,   // [T*B][4H] = x@Wx + b
    const float*  __restrict__ xwr,   // [T*B][128] = x@Wrx (cols>=100 are 0)
    const __bf16* __restrict__ WhT,   // [4H][H]
    const __bf16* __restrict__ WrhT,  // [128][H] (rows>=100 are 0)
    const __bf16* __restrict__ WdT,   // [H][128] (cols>=100 are 0)
    const float*  __restrict__ c0,    // [B][H]
    __bf16* __restrict__ hbf,         // [B][H]   recurrent h (bf16)
    __bf16* __restrict__ dbf,         // [B][128] recurrent d (bf16)
    float*  __restrict__ dfp,         // [B][128] recurrent d (f32)
    float*  __restrict__ out,         // outputs ++ h ++ c ++ d
    unsigned* __restrict__ bar)
{
    __shared__ float ge[2][4][BATCH][16];  // [K-half][gate type] partials, 32KB
    __shared__ float rbuf[2][BATCH][16];   // reading-gate partials, 8KB
    __shared__ float ddl[BATCH][16];       // d_new @ Wd (pre-tanh), 4KB
    __shared__ float cl[BATCH][16];        // cell state owned by this WG, 4KB

    const int j    = blockIdx.x;
    const int n0   = j * 16;
    const int w    = threadIdx.x >> 5;
    const int gt   = w & 3;            // gate type i/f/g/o (or m-subtile for r)
    const int kh   = w >> 2;           // K-half: 0 -> [0,512), 1 -> [512,1024)
    const int lane = threadIdx.x & 31;
    const int mofs = (lane >> 4) << 3;
    const int col  = lane & 15;
    const int kbase = kh * (HDIM / 2);

    const size_t offH = (size_t)T_STEPS * BATCH * HDIM;
    const size_t offC = offH + (size_t)BATCH * HDIM;
    const size_t offD = offC + (size_t)BATCH * HDIM;

    // init private cell state (256 threads x 4 elements)
#pragma unroll
    for (int e = 0; e < 4; ++e) {
        int idx = threadIdx.x * 4 + e;
        int m = idx >> 4, cc = idx & 15;
        cl[m][cc] = c0[(size_t)m * HDIM + n0 + cc];
    }

    for (int t = 0; t < T_STEPS; ++t) {
        grid_barrier(&bar[0], &bar[1]);   // h_t, d_t globally visible

        // ---- gates: partial acc = h_t[:, K-half] @ Wh[K-half, gt*H+n0 : +16]
        const int gcol0 = gt * HDIM + n0;
        v8f acc[4];
#pragma unroll
        for (int mi = 0; mi < 4; ++mi) acc[mi] = vzero8();
#pragma unroll 4
        for (int k0 = 0; k0 < HDIM / 2; k0 += 32) {
            v16bf bf = load_frag(WhT, HDIM, gcol0, kbase + k0);
#pragma unroll
            for (int mi = 0; mi < 4; ++mi) {
                v16bf af = load_frag(hbf, HDIM, mi * 16, kbase + k0);
                acc[mi] = __builtin_amdgcn_wmma_f32_16x16x32_bf16(
                    false, af, false, bf, (short)0, acc[mi], false, false);
            }
        }
#pragma unroll
        for (int mi = 0; mi < 4; ++mi) {
#pragma unroll
            for (int v = 0; v < 8; ++v) {
                int m = mi * 16 + v + mofs;
                ge[kh][gt][m][col] = acc[mi][v];
            }
        }

        // ---- reading gate partial: wave gt owns m-subtile gt*16, K-half kh
        if (j < 7) {
            const int dcol0 = j * 16;
            const int m0 = gt * 16;
            v8f racc = vzero8();
#pragma unroll 4
            for (int k0 = 0; k0 < HDIM / 2; k0 += 32) {
                v16bf bf = load_frag(WrhT, HDIM, dcol0, kbase + k0);
                v16bf af = load_frag(hbf, HDIM, m0, kbase + k0);
                racc = __builtin_amdgcn_wmma_f32_16x16x32_bf16(
                    false, af, false, bf, (short)0, racc, false, false);
            }
#pragma unroll
            for (int v = 0; v < 8; ++v) {
                rbuf[kh][m0 + v + mofs][col] = racc[v];
            }
        }
        __syncthreads();   // rbuf partials complete

        // ---- finalize r, update d (f32 master + bf16 operand copy)
        if (j < 7) {
            const float* xwrt = xwr + (size_t)t * BATCH * DPAD;
#pragma unroll
            for (int e = 0; e < 4; ++e) {
                int idx = threadIdx.x * 4 + e;
                int m = idx >> 4, cc = idx & 15;
                int dc = j * 16 + cc;
                float rv = sigmf(xwrt[(size_t)m * DPAD + dc] +
                                 0.5f * (rbuf[0][m][cc] + rbuf[1][m][cc]));
                float dn = rv * dfp[(size_t)m * DPAD + dc];
                dfp[(size_t)m * DPAD + dc] = dn;
                dbf[(size_t)m * DPAD + dc] = (__bf16)dn;
                if (t == T_STEPS - 1 && dc < DDIM)
                    out[offD + (size_t)m * DDIM + dc] = dn;
            }
        }

        grid_barrier(&bar[0], &bar[1]);   // d_new globally visible

        // ---- dd = d_new @ Wd[:, n0:n0+16]  (K padded to 128; waves 0..3)
        if (w < 4) {
            const int m0 = w * 16;
            v8f dacc = vzero8();
#pragma unroll
            for (int k0 = 0; k0 < DPAD; k0 += 32) {
                v16bf bf = load_frag(WdT, DPAD, n0, k0);
                v16bf af = load_frag(dbf, DPAD, m0, k0);
                dacc = __builtin_amdgcn_wmma_f32_16x16x32_bf16(
                    false, af, false, bf, (short)0, dacc, false, false);
            }
#pragma unroll
            for (int v = 0; v < 8; ++v) {
                ddl[m0 + v + mofs][col] = dacc[v];
            }
        }
        __syncthreads();

        // ---- elementwise fuse: sum K-half partials + xwx, c/h update
        const float* xwxt = xwx + (size_t)t * BATCH * H4;
#pragma unroll
        for (int e = 0; e < 4; ++e) {
            int idx = threadIdx.x * 4 + e;
            int m = idx >> 4, cc = idx & 15;
            size_t xb = (size_t)m * H4 + n0 + cc;
            float pi = ge[0][0][m][cc] + ge[1][0][m][cc] + xwxt[xb];
            float pf = ge[0][1][m][cc] + ge[1][1][m][cc] + xwxt[xb + HDIM];
            float pg = ge[0][2][m][cc] + ge[1][2][m][cc] + xwxt[xb + 2 * HDIM];
            float po = ge[0][3][m][cc] + ge[1][3][m][cc] + xwxt[xb + 3 * HDIM];
            float iv = sigmf(pi);
            float fv = sigmf(pf);
            float gv = tanhf(pg);
            float ov = sigmf(po);
            float cn = fv * cl[m][cc] + iv * gv + tanhf(ddl[m][cc]);
            float hn = ov * tanhf(cn);
            cl[m][cc] = cn;
            int hcol = n0 + cc;
            out[(size_t)t * BATCH * HDIM + (size_t)m * HDIM + hcol] = hn;
            hbf[(size_t)m * HDIM + hcol] = (__bf16)hn;
            if (t == T_STEPS - 1) {
                out[offH + (size_t)m * HDIM + hcol] = hn;
                out[offC + (size_t)m * HDIM + hcol] = cn;
            }
        }
        __syncthreads();
    }
}

// ---------------------------------------------------------------------------
// Host launcher
// ---------------------------------------------------------------------------

extern "C" void kernel_launch(void* const* d_in, const int* in_sizes, int n_in,
                              void* d_out, int out_size, void* d_ws, size_t ws_size,
                              hipStream_t stream) {
    (void)in_sizes; (void)n_in; (void)out_size; (void)ws_size;

    const float* x   = (const float*)d_in[0];  // [T,B,I]
    const float* h0  = (const float*)d_in[1];  // [B,H]
    const float* c0  = (const float*)d_in[2];  // [B,H]
    const float* d0  = (const float*)d_in[3];  // [B,D]
    const float* Wx  = (const float*)d_in[4];  // [I,4H]
    const float* Wh  = (const float*)d_in[5];  // [H,4H]
    const float* b   = (const float*)d_in[6];  // [4H]
    const float* Wrx = (const float*)d_in[7];  // [I,D]
    const float* Wrh = (const float*)d_in[8];  // [H,D]
    const float* Wd  = (const float*)d_in[9];  // [D,H]
    float* out = (float*)d_out;

    // Workspace carve-up
    char* wsp = (char*)d_ws;
    auto carve = [&](size_t bytes) {
        void* p = (void*)wsp;
        wsp += (bytes + 255) & ~(size_t)255;
        return p;
    };
    const size_t MROWS = (size_t)T_STEPS * BATCH;          // 16384
    __bf16* xbf  = (__bf16*)carve(MROWS * IDIM * 2);       // [T*B][I]
    __bf16* WxT  = (__bf16*)carve((size_t)H4 * IDIM * 2);  // [4H][I]
    __bf16* WhT  = (__bf16*)carve((size_t)H4 * HDIM * 2);  // [4H][H]
    __bf16* WrxT = (__bf16*)carve((size_t)DPAD * IDIM * 2);// [128][I]
    __bf16* WrhT = (__bf16*)carve((size_t)DPAD * HDIM * 2);// [128][H]
    __bf16* WdT  = (__bf16*)carve((size_t)HDIM * DPAD * 2);// [H][128]
    float*  xwx  = (float*)carve(MROWS * H4 * 4);          // [T*B][4H]
    float*  xwr  = (float*)carve(MROWS * DPAD * 4);        // [T*B][128]
    __bf16* hbf  = (__bf16*)carve((size_t)BATCH * HDIM * 2);
    __bf16* dbf  = (__bf16*)carve((size_t)BATCH * DPAD * 2);
    float*  dfp  = (float*)carve((size_t)BATCH * DPAD * 4);
    unsigned* bar = (unsigned*)carve(256);

    // 1) convert x to bf16 (row-major, A-operand layout)
    {
        int n = (int)(MROWS * IDIM);
        cvt_bf16<<<(n + 255) / 256, 256, 0, stream>>>(x, xbf, n);
    }
    // 2) transpose-convert all weight matrices into [n][k] bf16 (zero padded)
    {
        int n;
        n = H4 * IDIM;
        transpose_cvt<<<(n + 255) / 256, 256, 0, stream>>>(Wx,  WxT,  IDIM, H4,   H4,   IDIM);
        n = H4 * HDIM;
        transpose_cvt<<<(n + 255) / 256, 256, 0, stream>>>(Wh,  WhT,  HDIM, H4,   H4,   HDIM);
        n = DPAD * IDIM;
        transpose_cvt<<<(n + 255) / 256, 256, 0, stream>>>(Wrx, WrxT, IDIM, DDIM, DPAD, IDIM);
        n = DPAD * HDIM;
        transpose_cvt<<<(n + 255) / 256, 256, 0, stream>>>(Wrh, WrhT, HDIM, DDIM, DPAD, HDIM);
        n = HDIM * DPAD;
        transpose_cvt<<<(n + 255) / 256, 256, 0, stream>>>(Wd,  WdT,  DDIM, HDIM, HDIM, DPAD);
    }
    // 3) init recurrent state + barrier
    {
        int n = BATCH * HDIM;
        init_state<<<(n + 255) / 256, 256, 0, stream>>>(h0, d0, hbf, dbf, dfp, bar);
    }
    // 4) parallel precompute: xwx = x@Wx + b   [16384 x 4096, K=512]
    {
        dim3 grid(H4 / 64, (unsigned)(MROWS / 64));
        gemm_bf16_tn<<<grid, 128, 0, stream>>>(xbf, WxT, b, xwx,
                                               (int)MROWS, H4, IDIM);
    }
    // 5) parallel precompute: xwr = x@Wrx      [16384 x 128, K=512]
    {
        dim3 grid(DPAD / 64, (unsigned)(MROWS / 64));
        gemm_bf16_tn<<<grid, 128, 0, stream>>>(xbf, WrxT, nullptr, xwr,
                                               (int)MROWS, DPAD, IDIM);
    }
    // 6) persistent recurrence over T steps
    sclstm_recur<<<NWG, 256, 0, stream>>>(xwx, xwr, WhT, WrhT, WdT, c0,
                                          hbf, dbf, dfp, out, bar);
}
